// MoDEChameleonMLP_37898791420361
// MI455X (gfx1250) — compile-verified, matrix operand order checked by reading, not run on hardware
//
#include <hip/hip_runtime.h>
#include <hip/hip_bf16.h>
#include <cstdint>
#include <cstddef>

// ---------------------------------------------------------------------------
// MoDE Chameleon MLP for MI455X (gfx1250, wave32, WMMA).
// B=2,S=2048 -> N=4096 tokens, H=4096, I=11008, E=4, R=8, SCALE=2.
// bf16 WMMA GEMMs with LDS tiles staged by GLOBAL_LOAD_ASYNC_TO_LDS_B128
// (ASYNCcnt) when the toolchain exposes the builtin; adapter deltas folded in
// as a rank-64 extra-K WMMA contribution; gate/up/SiLU fused in one pass.
// ---------------------------------------------------------------------------

typedef __attribute__((ext_vector_type(16))) __bf16 v16bf;
typedef __attribute__((ext_vector_type(8)))  __bf16 v8bf;
typedef __attribute__((ext_vector_type(8)))  float  v8f;

static constexpr float kScale = 2.0f;          // lora_alpha / rank
static constexpr int BM = 128, BN = 128, BK = 32;
static constexpr int STR = 40;                 // LDS row stride in halves (80B, 16B aligned)

// ---------------- async global->LDS path (CDNA5) --------------------------

#if defined(__has_builtin)
#if __has_builtin(__builtin_amdgcn_global_load_async_to_lds_b128) && \
    __has_builtin(__builtin_amdgcn_s_wait_asynccnt)
#define MODE_ASYNC_LDS 1
#endif
#endif
#ifndef MODE_ASYNC_LDS
#define MODE_ASYNC_LDS 0
#endif

#if MODE_ASYNC_LDS
// Builtin signature (from hipcc diagnostic): param0 is
// 'int __attribute__((vector_size(16))) addrspace(1)*', param1 addrspace(3).
typedef int v4i_vs __attribute__((vector_size(16)));
typedef __attribute__((address_space(1))) v4i_vs gv4i;
typedef __attribute__((address_space(3))) v4i_vs lv4i;
// Per-lane 16B global->LDS DMA; tracked by ASYNCcnt, no VGPR staging.
// Generic LDS pointer low 32 bits == LDS byte offset (ISA section 10.2).
__device__ __forceinline__ void async_cp16(void* lds, const void* g) {
  __builtin_amdgcn_global_load_async_to_lds_b128(
      (gv4i*)(uintptr_t)g, (lv4i*)(uint32_t)(uintptr_t)lds, 0, 0);
}
__device__ __forceinline__ void wait_async0() {
  __builtin_amdgcn_s_wait_asynccnt(0);
}
#endif

// ---------------- WMMA helpers -------------------------------------------

__device__ __forceinline__ v8f wmma_bf16(v16bf a, v16bf b, v8f c) {
  // D = A(16x32) * B(32x16) + C(16x16), f32 accumulate
  return __builtin_amdgcn_wmma_f32_16x16x32_bf16(
      /*neg_a=*/false, a, /*neg_b=*/false, b,
      /*c_mod=*/(short)0, c, /*reuse_a=*/false, /*reuse_b=*/false);
}

// A/B fragment from an LDS tile (row-major, STR halves per row).
// ISA 16-bit operand layout: lanes 0-15 hold K=0..7,16..23 of row (lane),
// lanes 16-31 hold K=8..15,24..31 of row (lane-16).
__device__ __forceinline__ v16bf frag_lds(const __bf16* tile, int lane) {
  const int l = lane & 15, hk = (lane >> 4) & 1;
  const __bf16* p = tile + l * STR + hk * 8;
  union { v16bf v; v8bf h[2]; } u;
  u.h[0] = *(const v8bf*)(p);        // K 0..7   (or 8..15)
  u.h[1] = *(const v8bf*)(p + 16);   // K 16..23 (or 24..31)
  return u.v;
}

// Same layout straight from global memory with row stride 64 halves
// (rank-64 adapter tail: coef[N,64] and Bcomb[out,64]).
__device__ __forceinline__ v16bf frag_g64(const __bf16* row0, int lane, int koff) {
  const int l = lane & 15, hk = (lane >> 4) & 1;
  const __bf16* p = row0 + (size_t)l * 64 + koff + hk * 8;
  union { v16bf v; v8bf h[2]; } u;
  u.h[0] = *(const v8bf*)(p);
  u.h[1] = *(const v8bf*)(p + 16);
  return u.v;
}

// ---------------- elementwise f32 -> bf16 conversion ----------------------

__global__ __launch_bounds__(256) void mode_cvt_bf16(
    const float* __restrict__ src, __bf16* __restrict__ dst, long long n4) {
  long long i = (long long)blockIdx.x * blockDim.x + threadIdx.x;
  if (i >= n4) return;
  const float4 f = *(const float4*)(src + i * 4);
  union { uint2 u; __bf16 h[4]; } w;
  w.h[0] = (__bf16)f.x; w.h[1] = (__bf16)f.y;
  w.h[2] = (__bf16)f.z; w.h[3] = (__bf16)f.w;
  *(uint2*)((char*)dst + i * 8) = w.u;
}

// ---------------- combined adapter basis ----------------------------------
// out[i][j] : j<32 -> tmB[e=j/8][i][r=j%8], j in [32,40) -> vB[i][j-32], else 0.

__global__ __launch_bounds__(256) void mode_bcomb(
    const float* __restrict__ tmB,   // [E, rows, R]
    const float* __restrict__ vB,    // [rows, R]
    int rows, __bf16* __restrict__ outB) {
  long long idx = (long long)blockIdx.x * blockDim.x + threadIdx.x;
  if (idx >= (long long)rows * 64) return;
  const int i = (int)(idx >> 6), j = (int)(idx & 63);
  float v = 0.0f;
  if (j < 32) {
    const int e = j >> 3, r = j & 7;
    v = tmB[((size_t)e * rows + i) * 8 + r];
  } else if (j < 40) {
    v = vB[(size_t)i * 8 + (j - 32)];
  }
  outB[idx] = (__bf16)v;
}

// ---------------- per-token adapter coefficients --------------------------
// One block (8 waves) per token: 44 length-K dot products (4 router logits,
// 32 MoE A-rows, 8 V-adapter A-rows), wave32 shuffle reduction, then
// softmax + mask select into coef[n][0..63] (cols 40..63 forced to zero).

template <int BF>
__global__ __launch_bounds__(256) void mode_coef(
    const float*  __restrict__ Xf,     // f32 input (BF=0)
    const __bf16* __restrict__ Xb,     // bf16 input (BF=1)
    int K,
    const float* __restrict__ router,  // [4,K]
    const float* __restrict__ tmA,     // [4,8,K] flattened
    const float* __restrict__ vA,      // [8,K]
    const int*   __restrict__ mask,    // [N]
    __bf16* __restrict__ coef) {       // [N,64]
  __shared__ float sums[44];
  const int n = blockIdx.x;
  const int tid = threadIdx.x, lane = tid & 31, wave = tid >> 5;

  for (int j = wave; j < 44; j += 8) {
    const float* wrow = (j < 4)  ? router + (size_t)j * K
                      : (j < 36) ? tmA + (size_t)(j - 4) * K
                                 : vA  + (size_t)(j - 36) * K;
    float acc = 0.0f;
    for (int k = lane; k < K; k += 32) {
      const float xv = BF ? (float)Xb[(size_t)n * K + k] : Xf[(size_t)n * K + k];
      acc += xv * wrow[k];
    }
#pragma unroll
    for (int off = 16; off > 0; off >>= 1) acc += __shfl_xor(acc, off);
    if (lane == 0) sums[j] = acc;
  }
  __syncthreads();

  if (tid == 0) {
    float o[64];
#pragma unroll
    for (int j = 0; j < 64; ++j) o[j] = 0.0f;
    if (mask[n] != 0) {                       // image token -> V-adapter LoRA
      for (int r = 0; r < 8; ++r) o[32 + r] = kScale * sums[36 + r];
    } else {                                  // text token -> dense-softmax MoE
      float mx = sums[0];
      for (int e = 1; e < 4; ++e) mx = fmaxf(mx, sums[e]);
      float l[4], s = 0.0f;
      for (int e = 0; e < 4; ++e) { l[e] = __expf(sums[e] - mx); s += l[e]; }
      const float inv = kScale / s;
      for (int e = 0; e < 4; ++e)
        for (int r = 0; r < 8; ++r)
          o[e * 8 + r] = inv * l[e] * sums[4 + e * 8 + r];
    }
    for (int j = 0; j < 64; ++j) coef[(size_t)n * 64 + j] = (__bf16)o[j];
  }
}

// ---------------- main WMMA GEMM ------------------------------------------
// C[M,Nn] = A[M,K] * W[Nn,K]^T + coef[M,64] * Bcomb[Nn,64]^T
// DUAL=1: gate & up share the A tile; epilogue silu(gate)*up -> bf16 inter.
// DUAL=0: single matrix; epilogue stores f32.
// Workgroup: 256 threads (8 waves, 2x4), tile 128x128, wave tile 64x32.
// Async build: double-buffered LDS fed by GLOBAL_LOAD_ASYNC_TO_LDS_B128.
// Fallback:    single-buffer synchronous staging (no long-lived regs).

template <int DUAL>
__global__ __launch_bounds__(256, 1) void mode_gemm(
    const __bf16* __restrict__ Am,    // [M,K] row-major
    const __bf16* __restrict__ W0,    // [Nn,K] row-major
    const __bf16* __restrict__ W1,    // [Nn,K] (DUAL only)
    const __bf16* __restrict__ c0,    // [M,64]
    const __bf16* __restrict__ c1,    // [M,64] (DUAL only)
    const __bf16* __restrict__ bc0,   // [Nn,64]
    const __bf16* __restrict__ bc1,   // [Nn,64] (DUAL only)
    int K, int Nn,
    __bf16* __restrict__ interOut,    // DUAL=1 output
    float*  __restrict__ fOut) {      // DUAL=0 output
#if MODE_ASYNC_LDS
  constexpr int NBUF = 2;
#else
  constexpr int NBUF = 1;
#endif
  __shared__ __align__(16) __bf16 sm[(DUAL ? 3 : 2) * NBUF * BM * STR];
  __bf16* As  = sm;                               // [NBUF][BM][STR]
  __bf16* Bs0 = sm + NBUF * BM * STR;             // [NBUF][BN][STR]
  __bf16* Bs1 = sm + 2 * NBUF * BM * STR;         // [NBUF][BN][STR] (DUAL)

  const int tid  = threadIdx.x;
  const int lane = tid & 31;
  const int wave = tid >> 5;
  const int wm   = wave >> 2;          // 0..1 -> 64 rows each
  const int wn   = wave & 3;           // 0..3 -> 32 cols each
  const int m0 = blockIdx.y * BM;
  const int n0 = blockIdx.x * BN;

  // Stage one 128x32 bf16 tile (512 x 16B) from global into an LDS buffer.
  auto stage_one = [&](const __bf16* src, int row0, int kk, __bf16* dstbase) {
#pragma unroll
    for (int i = 0; i < 2; ++i) {
      const int idx = tid + i * 256, row = idx >> 2, seg = idx & 3;
      const __bf16* g = src + (size_t)(row0 + row) * K + kk + seg * 8;
      __bf16* l = dstbase + row * STR + seg * 8;
#if MODE_ASYNC_LDS
      async_cp16(l, g);
#else
      *(uint4*)l = *(const uint4*)g;
#endif
    }
  };
  auto stage = [&](int buf, int kk) {
    stage_one(Am, m0, kk, As + buf * BM * STR);
    stage_one(W0, n0, kk, Bs0 + buf * BM * STR);
    if (DUAL) stage_one(W1, n0, kk, Bs1 + buf * BM * STR);
  };

  v8f accG[4][2] = {};
  v8f accU[4][2] = {};

  auto compute = [&](int buf) {
    const __bf16* Ab  = As  + buf * BM * STR + (wm * 64) * STR;
    const __bf16* B0b = Bs0 + buf * BM * STR + (wn * 32) * STR;
    const __bf16* B1b = Bs1 + buf * BM * STR + (wn * 32) * STR;
    v16bf af[4];
#pragma unroll
    for (int mf = 0; mf < 4; ++mf) af[mf] = frag_lds(Ab + mf * 16 * STR, lane);
#pragma unroll
    for (int nf = 0; nf < 2; ++nf) {
      const v16bf b0 = frag_lds(B0b + nf * 16 * STR, lane);
#pragma unroll
      for (int mf = 0; mf < 4; ++mf) accG[mf][nf] = wmma_bf16(af[mf], b0, accG[mf][nf]);
      if (DUAL) {
        const v16bf b1 = frag_lds(B1b + nf * 16 * STR, lane);
#pragma unroll
        for (int mf = 0; mf < 4; ++mf) accU[mf][nf] = wmma_bf16(af[mf], b1, accU[mf][nf]);
      }
    }
  };

  const int KC = K / BK;
#if MODE_ASYNC_LDS
  // software pipeline: DMA chunk kc+1 while WMMAs chew on chunk kc
  stage(0, 0);
  wait_async0();
  __syncthreads();
  for (int kc = 0; kc < KC; ++kc) {
    const int p = kc & 1;
    if (kc + 1 < KC) stage(1 - p, (kc + 1) * BK);   // async, no VGPR staging
    compute(p);
    if (kc + 1 < KC) {
      wait_async0();                                // my copies into buf 1-p done
      __syncthreads();                              // everyone's copies done
    }
  }
#else
  for (int kc = 0; kc < KC; ++kc) {
    if (kc) __syncthreads();                        // done reading buffer
    stage(0, kc * BK);
    __syncthreads();
    compute(0);
  }
#endif

  // rank-64 adapter tail: two extra k-chunks straight from global memory
#pragma unroll
  for (int t = 0; t < 2; ++t) {
    const int kk = t * 32;
    v16bf afg[4];
#pragma unroll
    for (int mf = 0; mf < 4; ++mf)
      afg[mf] = frag_g64(c0 + (size_t)(m0 + wm * 64 + mf * 16) * 64, lane, kk);
#pragma unroll
    for (int nf = 0; nf < 2; ++nf) {
      const v16bf b0 = frag_g64(bc0 + (size_t)(n0 + wn * 32 + nf * 16) * 64, lane, kk);
#pragma unroll
      for (int mf = 0; mf < 4; ++mf) accG[mf][nf] = wmma_bf16(afg[mf], b0, accG[mf][nf]);
    }
    if (DUAL) {
      v16bf afu[4];
#pragma unroll
      for (int mf = 0; mf < 4; ++mf)
        afu[mf] = frag_g64(c1 + (size_t)(m0 + wm * 64 + mf * 16) * 64, lane, kk);
#pragma unroll
      for (int nf = 0; nf < 2; ++nf) {
        const v16bf b1 = frag_g64(bc1 + (size_t)(n0 + wn * 32 + nf * 16) * 64, lane, kk);
#pragma unroll
        for (int mf = 0; mf < 4; ++mf) accU[mf][nf] = wmma_bf16(afu[mf], b1, accU[mf][nf]);
      }
    }
  }

  // epilogue.  C layout: VGPR r -> row = r + 8*(lane>=16); col = lane&15.
  const int colBase = n0 + wn * 32 + (lane & 15);
  const int rowHi   = ((lane >> 4) & 1) * 8;
#pragma unroll
  for (int mf = 0; mf < 4; ++mf) {
#pragma unroll
    for (int nf = 0; nf < 2; ++nf) {
      const int col = colBase + nf * 16;
#pragma unroll
      for (int r = 0; r < 8; ++r) {
        const int row = m0 + wm * 64 + mf * 16 + r + rowHi;
        if (DUAL) {
          const float g = accG[mf][nf][r];
          const float u = accU[mf][nf][r];
          const float s = g / (1.0f + __expf(-g));    // silu(g)
          interOut[(size_t)row * Nn + col] = (__bf16)(s * u);
        } else {
          fOut[(size_t)row * Nn + col] = accG[mf][nf][r];
        }
      }
    }
  }
}

// ---------------------------------------------------------------------------

extern "C" void kernel_launch(void* const* d_in, const int* in_sizes, int n_in,
                              void* d_out, int out_size, void* d_ws, size_t ws_size,
                              hipStream_t stream) {
  (void)in_sizes; (void)n_in; (void)out_size; (void)ws_size;
  constexpr int N = 2 * 2048, H = 4096, I = 11008;

  const float* x   = (const float*)d_in[0];
  const int*   msk = (const int*)d_in[1];     // image_mask (0/1)
  const float* Wg  = (const float*)d_in[2];
  const float* Wu  = (const float*)d_in[3];
  const float* Wd  = (const float*)d_in[4];
  const float* vgA = (const float*)d_in[5];
  const float* vgB = (const float*)d_in[6];
  const float* vuA = (const float*)d_in[7];
  const float* vuB = (const float*)d_in[8];
  const float* vdA = (const float*)d_in[9];
  const float* vdB = (const float*)d_in[10];
  const float* rg  = (const float*)d_in[11];
  const float* tgA = (const float*)d_in[12];
  const float* tgB = (const float*)d_in[13];
  const float* ru  = (const float*)d_in[14];
  const float* tuA = (const float*)d_in[15];
  const float* tuB = (const float*)d_in[16];
  const float* rd  = (const float*)d_in[17];
  const float* tdA = (const float*)d_in[18];
  const float* tdB = (const float*)d_in[19];
  float* out = (float*)d_out;

  // workspace carve-out (~400 MB)
  char* ws = (char*)d_ws;
  size_t off = 0;
  auto alloc = [&](size_t bytes) -> char* {
    char* p = ws + off;
    off = (off + bytes + 255) & ~(size_t)255;
    return p;
  };
  __bf16* xb     = (__bf16*)alloc((size_t)N * H * 2);
  __bf16* Wgb    = (__bf16*)alloc((size_t)I * H * 2);
  __bf16* Wub    = (__bf16*)alloc((size_t)I * H * 2);
  __bf16* Wdb    = (__bf16*)alloc((size_t)H * I * 2);
  __bf16* interb = (__bf16*)alloc((size_t)N * I * 2);
  __bf16* coefG  = (__bf16*)alloc((size_t)N * 64 * 2);
  __bf16* coefU  = (__bf16*)alloc((size_t)N * 64 * 2);
  __bf16* coefD  = (__bf16*)alloc((size_t)N * 64 * 2);
  __bf16* BcG    = (__bf16*)alloc((size_t)I * 64 * 2);
  __bf16* BcU    = (__bf16*)alloc((size_t)I * 64 * 2);
  __bf16* BcD    = (__bf16*)alloc((size_t)H * 64 * 2);

  auto cvt = [&](const float* s, __bf16* d, long long n) {
    const long long n4 = n / 4;
    mode_cvt_bf16<<<(unsigned)((n4 + 255) / 256), 256, 0, stream>>>(s, d, n4);
  };

  // 1) f32 -> bf16 (one pass; bf16 Wg+Wu = 180MB ~ lives in the 192MB L2)
  cvt(x,  xb,  (long long)N * H);
  cvt(Wg, Wgb, (long long)I * H);
  cvt(Wu, Wub, (long long)I * H);
  cvt(Wd, Wdb, (long long)H * I);

  // 2) combined adapter bases [out,64]
  mode_bcomb<<<(I * 64 + 255) / 256, 256, 0, stream>>>(tgB, vgB, I, BcG);
  mode_bcomb<<<(I * 64 + 255) / 256, 256, 0, stream>>>(tuB, vuB, I, BcU);
  mode_bcomb<<<(H * 64 + 255) / 256, 256, 0, stream>>>(tdB, vdB, H, BcD);

  // 3) per-token coefficients for gate/up (K = H, f32 input)
  mode_coef<0><<<N, 256, 0, stream>>>(x, nullptr, H, rg, tgA, vgA, msk, coefG);
  mode_coef<0><<<N, 256, 0, stream>>>(x, nullptr, H, ru, tuA, vuA, msk, coefU);

  // 4) fused gate/up GEMM + adapters + silu(gate)*up -> inter (bf16)
  mode_gemm<1><<<dim3(I / BN, N / BM), 256, 0, stream>>>(
      xb, Wgb, Wub, coefG, coefU, BcG, BcU, H, I, interb, nullptr);

  // 5) down-path coefficients on inter (K = I, bf16 input)
  mode_coef<1><<<N, 256, 0, stream>>>(nullptr, interb, I, rd, tdA, vdA, msk, coefD);

  // 6) down GEMM + adapters -> f32 output
  mode_gemm<0><<<dim3(H / BN, N / BM), 256, 0, stream>>>(
      interb, Wdb, nullptr, coefD, nullptr, BcD, nullptr, I, H, nullptr, out);
}
